// ObjectRelationModuleWithAttention_43422119363015
// MI455X (gfx1250) — matrix operand order repaired and checked by reading, not compiled, not run on hardware
//
#include <hip/hip_runtime.h>
#include <hip/hip_bf16.h>

// ---------------------------------------------------------------------------
// Shapes (compile-time constants from the reference)
// ---------------------------------------------------------------------------
constexpr int cB  = 16;      // batch
constexpr int cN  = 256;     // rois
constexpr int cF  = 1024;    // feat dim
constexpr int cG  = 16;      // groups
constexpr int cE  = 64;      // pos-emb dim
constexpr int cD  = 1024;    // qkv dim
constexpr int cDG = 64;      // per-group dim
constexpr int cBN = cB * cN; // 4096

typedef __attribute__((ext_vector_type(16))) _Float16 v16h;
typedef __attribute__((ext_vector_type(8)))  float    v8f;

// ---------------------------------------------------------------------------
// WMMA helper: D = A(16x32 f16) * B(32x16 f16) + C(16x16 f32)
// ---------------------------------------------------------------------------
__device__ __forceinline__ v8f wmma_f16(v16h a, v16h b, v8f c) {
  return __builtin_amdgcn_wmma_f32_16x16x32_f16(
      /*neg_a=*/false, a, /*neg_b=*/false, b,
      /*c_mod=*/(short)0, c, /*reuse_a=*/false, /*reuse_b=*/false);
}

// A fragment: source row-major [M, lda], tile origin (m0, k0), 16x32 tile.
// Layout: lane<16 -> M=lane, K = k0..k0+15;
//         lane>=16 -> M=lane-16, K = k0+16..k0+31 (matches load_b_frag).
__device__ __forceinline__ v16h load_a_frag(const _Float16* A, int lda,
                                            int m0, int k0, int lane) {
  int row = m0 + (lane & 15);
  int kb  = k0 + ((lane >> 4) << 4);
  const _Float16* p = A + (size_t)row * lda + kb;
  v16h f;
#pragma unroll
  for (int h = 0; h < 8; ++h) f[h] = p[h];
#pragma unroll
  for (int h = 0; h < 8; ++h) f[8 + h] = p[8 + h];
  return f;
}

// B fragment: we always compute C = A * Bsrc^T where Bsrc is row-major [N, ldb]
// (i.e. frag element (k, n) = Bsrc[n][k]).  Contiguous k order, same as load_a_frag: n = lane&15,
// K = k0 + h + 16*(lane>=16)  -> one contiguous 16-half run per lane.
__device__ __forceinline__ v16h load_b_frag(const _Float16* Bsrc, int ldb,
                                            int n0, int k0, int lane) {
  int col = n0 + (lane & 15);
  int kb  = k0 + ((lane >> 4) << 4);
  const _Float16* p = Bsrc + (size_t)col * ldb + kb;
  v16h f;
#pragma unroll
  for (int h = 0; h < 16; ++h) f[h] = p[h];
  return f;
}

// ---------------------------------------------------------------------------
// f32 -> f16 conversion (roi feats + weights)
// ---------------------------------------------------------------------------
__global__ void cvt_f32_to_f16(const float* __restrict__ in,
                               _Float16* __restrict__ out, int n) {
  int i = blockIdx.x * blockDim.x + threadIdx.x;
  if (i < n) out[i] = (_Float16)in[i];
}

// ---------------------------------------------------------------------------
// Position bias: stream pe ONCE, emit log(relu(pe . Wp[g] + bp[g])) for all
// 16 groups as f16 [B][G][N][N].  Block = 256 threads = one (b, i) pair;
// thread j keeps its 256-byte pe[b, j, i, :] vector in registers and reuses
// it across all 16 groups (Wp/bp cached in LDS).
// ---------------------------------------------------------------------------
__global__ void pbias_kernel(const float* __restrict__ pe,
                             const float* __restrict__ Wp,
                             const float* __restrict__ bp,
                             _Float16* __restrict__ pbias) {
  __shared__ float wp_s[cG * cE];   // 4 KB
  __shared__ float bp_s[cG];
  const int j = threadIdx.x;        // 0..255
  const int i = blockIdx.x & 255;
  const int b = blockIdx.x >> 8;

  for (int t = threadIdx.x; t < cG * cE; t += blockDim.x) wp_s[t] = Wp[t];
  if (threadIdx.x < cG) bp_s[threadIdx.x] = bp[threadIdx.x];
  __syncthreads();

  // pe[b, j, i, :] -> registers (16 x float4)
  const float4* pev =
      reinterpret_cast<const float4*>(pe + (((size_t)b * cN + j) * cN + i) * cE);
  float pv[cE];
#pragma unroll
  for (int e4 = 0; e4 < cE / 4; ++e4) {
    float4 t = pev[e4];
    pv[4 * e4 + 0] = t.x;
    pv[4 * e4 + 1] = t.y;
    pv[4 * e4 + 2] = t.z;
    pv[4 * e4 + 3] = t.w;
  }

#pragma unroll 1
  for (int g = 0; g < cG; ++g) {
    float dot = bp_s[g];
#pragma unroll
    for (int e = 0; e < cE; ++e) dot += pv[e] * wp_s[g * cE + e];
    float lg = __logf(fmaxf(dot, 0.0f));
    pbias[(((size_t)(b * cG + g)) * cN + i) * cN + j] = (_Float16)lg;
  }
}

// ---------------------------------------------------------------------------
// QKV projection: Y = X(BNxF f16) @ W^T(DxF f16) + bias, stored per mode:
//   mode 0 (q): scaled by 1/sqrt(DG), layout [B][G][N][DG] f16
//   mode 1 (k):                       layout [B][G][N][DG] f16
//   mode 2 (v):                       layout [B][F][N]      f16 (transposed)
// One wave computes one 16x16 C tile; 4 waves per block.
// ---------------------------------------------------------------------------
__global__ void qkv_gemm(const _Float16* __restrict__ X,
                         const _Float16* __restrict__ W,
                         const float* __restrict__ bias,
                         _Float16* __restrict__ out,
                         int mode, float scale) {
  const int lane = threadIdx.x & 31;
  const int wid  = blockIdx.x * (blockDim.x >> 5) + (threadIdx.x >> 5);
  const int m0   = (wid >> 6) << 4;   // BN/16 = 256 row tiles
  const int n0   = (wid & 63) << 4;   // D/16  = 64  col tiles

  v8f acc = {};
#pragma unroll 4
  for (int k0 = 0; k0 < cF; k0 += 32) {
    v16h a = load_a_frag(X, cF, m0, k0, lane);
    v16h b = load_b_frag(W, cF, n0, k0, lane);  // W row-major [D, F]
    acc = wmma_f16(a, b, acc);
  }

  const int hi = lane >> 4;
  const int nl = lane & 15;
  const int n  = n0 + nl;
  const float bn = bias[n];
#pragma unroll
  for (int r = 0; r < 8; ++r) {
    int   m   = m0 + r + 8 * hi;
    float val = (acc[r] + bn) * scale;
    int b_idx = m >> 8;           // batch
    int i_idx = m & 255;          // roi
    size_t addr;
    if (mode == 2) {              // vT: [B][F][N]
      addr = ((size_t)b_idx * cF + n) * cN + i_idx;
    } else {                      // q/k: [B][G][N][DG]
      int g = n >> 6, dl = n & 63;
      addr = (((size_t)b_idx * cG + g) * cN + i_idx) * cDG + dl;
    }
    out[addr] = (_Float16)val;
  }
}

// ---------------------------------------------------------------------------
// Scores: per (b, g, i-tile of 16): qk via WMMA + precomputed log-bias,
// row softmax over j (256), store S as f16 [B][G][N][N].
// One wave per block.
// ---------------------------------------------------------------------------
__global__ void scores_kernel(const _Float16* __restrict__ qbuf,
                              const _Float16* __restrict__ kbuf,
                              const _Float16* __restrict__ pbias,
                              _Float16* __restrict__ sbuf) {
  const int lane = threadIdx.x;
  const int id   = blockIdx.x;
  const int it   = id & 15;
  const int g    = (id >> 4) & 15;
  const int b    = id >> 8;
  const int i0   = it * 16;

  const _Float16* qg = qbuf + ((size_t)(b * cG + g) * cN) * cDG;
  const _Float16* kg = kbuf + ((size_t)(b * cG + g) * cN) * cDG;

  v16h a0 = load_a_frag(qg, cDG, i0, 0, lane);
  v16h a1 = load_a_frag(qg, cDG, i0, 32, lane);

  float logit[16][8];
#pragma unroll
  for (int jt = 0; jt < 16; ++jt) {
    v16h b0 = load_b_frag(kg, cDG, jt * 16, 0, lane);
    v16h b1 = load_b_frag(kg, cDG, jt * 16, 32, lane);
    v8f acc = {};
    acc = wmma_f16(a0, b0, acc);
    acc = wmma_f16(a1, b1, acc);
#pragma unroll
    for (int r = 0; r < 8; ++r) logit[jt][r] = acc[r];
  }

  const int hi = lane >> 4;
  const int nl = lane & 15;

  // add precomputed log position bias (f16, [B][G][N][N])
  const _Float16* pb = pbias + ((size_t)(b * cG + g) * cN) * cN;
#pragma unroll
  for (int r = 0; r < 8; ++r) {
    int i = i0 + r + 8 * hi;
    const _Float16* prow = pb + (size_t)i * cN + nl;
#pragma unroll
    for (int jt = 0; jt < 16; ++jt)
      logit[jt][r] += (float)prow[jt * 16];
  }

  // softmax over j (16 lanes of this half x 16 jt values), then store f16
#pragma unroll 1
  for (int r = 0; r < 8; ++r) {
    float mx = -INFINITY;
#pragma unroll
    for (int jt = 0; jt < 16; ++jt) mx = fmaxf(mx, logit[jt][r]);
    for (int off = 8; off; off >>= 1) mx = fmaxf(mx, __shfl_xor(mx, off, 16));
    float sum = 0.0f;
#pragma unroll
    for (int jt = 0; jt < 16; ++jt) {
      float ev = __expf(logit[jt][r] - mx);
      logit[jt][r] = ev;
      sum += ev;
    }
    for (int off = 8; off; off >>= 1) sum += __shfl_xor(sum, off, 16);
    float inv = 1.0f / sum;
    int i = i0 + r + 8 * hi;
    _Float16* srow = sbuf + ((size_t)(b * cG + g) * cN + i) * cN;
#pragma unroll
    for (int jt = 0; jt < 16; ++jt)
      srow[jt * 16 + nl] = (_Float16)(logit[jt][r] * inv);
  }
}

// ---------------------------------------------------------------------------
// Output: per (b, g, i-tile of 16):
//   GEMM1: out_t = S(16x256) @ V  -> 16x1024 f16 tile staged in LDS
//   GEMM2: out   = out_t @ Wo[g]^T + bo -> 16x64 f32 slice of d_out
// 4 waves per block.
// ---------------------------------------------------------------------------
__global__ void out_kernel(const _Float16* __restrict__ sbuf,
                           const _Float16* __restrict__ vT,
                           const _Float16* __restrict__ Woh,
                           const float* __restrict__ bo,
                           float* __restrict__ out) {
  __shared__ _Float16 otile[16 * cF];   // 32 KB
  const int lane = threadIdx.x & 31;
  const int warp = threadIdx.x >> 5;
  const int it   = blockIdx.x & 15;
  const int g    = (blockIdx.x >> 4) & 15;
  const int b    = blockIdx.x >> 8;
  const int i0   = it * 16;
  const int hi   = lane >> 4;
  const int nl   = lane & 15;

  // GEMM1: A = S rows [16, 256], B = vT[b] ([F rows, N cols])
  const _Float16* S  = sbuf + ((size_t)(b * cG + g) * cN + i0) * cN;
  const _Float16* Vt = vT + (size_t)b * cF * cN;

  v16h af[8];
#pragma unroll
  for (int kk = 0; kk < 8; ++kk) af[kk] = load_a_frag(S, cN, 0, kk * 32, lane);

#pragma unroll 1
  for (int t = 0; t < 16; ++t) {
    int n0 = (warp * 16 + t) << 4;       // 0..1008
    v8f acc = {};
#pragma unroll
    for (int kk = 0; kk < 8; ++kk) {
      v16h bf = load_b_frag(Vt, cN, n0, kk * 32, lane);
      acc = wmma_f16(af[kk], bf, acc);
    }
#pragma unroll
    for (int r = 0; r < 8; ++r)
      otile[(r + 8 * hi) * cF + n0 + nl] = (_Float16)acc[r];
  }
  __syncthreads();

  // GEMM2: A = otile [16, 1024] (LDS), B = Wo[g] ([64 rows, 1024])
  const _Float16* Wog = Woh + (size_t)g * cDG * cF;
  const int n0b = warp << 4;             // 0,16,32,48
  v8f acc = {};
#pragma unroll 4
  for (int kk = 0; kk < 32; ++kk) {
    v16h a  = load_a_frag(otile, cF, 0, kk * 32, lane);
    v16h bf = load_b_frag(Wog, cF, n0b, kk * 32, lane);
    acc = wmma_f16(a, bf, acc);
  }
#pragma unroll
  for (int r = 0; r < 8; ++r) {
    int m = i0 + r + 8 * hi;
    int n = n0b + nl;
    out[((size_t)(b * cN + m)) * cD + g * cDG + n] = acc[r] + bo[g * cDG + n];
  }
}

// ---------------------------------------------------------------------------
// Host launcher
// ---------------------------------------------------------------------------
extern "C" void kernel_launch(void* const* d_in, const int* in_sizes, int n_in,
                              void* d_out, int out_size, void* d_ws, size_t ws_size,
                              hipStream_t stream) {
  const float* roi = (const float*)d_in[0];
  const float* pe  = (const float*)d_in[1];
  const float* Wq  = (const float*)d_in[2];
  const float* bq  = (const float*)d_in[3];
  const float* Wk  = (const float*)d_in[4];
  const float* bk  = (const float*)d_in[5];
  const float* Wv  = (const float*)d_in[6];
  const float* bv  = (const float*)d_in[7];
  const float* Wp  = (const float*)d_in[8];
  const float* bp  = (const float*)d_in[9];
  const float* Wo  = (const float*)d_in[10];
  const float* bo  = (const float*)d_in[11];
  float* out = (float*)d_out;

  // ---- workspace carve-up (all f16 buffers) ----
  _Float16* wsb = (_Float16*)d_ws;
  size_t off = 0;
  _Float16* Xh    = wsb + off; off += (size_t)cBN * cF;        // 4096x1024
  _Float16* Wqh   = wsb + off; off += (size_t)cD * cF;
  _Float16* Wkh   = wsb + off; off += (size_t)cD * cF;
  _Float16* Wvh   = wsb + off; off += (size_t)cD * cF;
  _Float16* Woh   = wsb + off; off += (size_t)cG * cDG * cF;
  _Float16* qbuf  = wsb + off; off += (size_t)cB * cG * cN * cDG;
  _Float16* kbuf  = wsb + off; off += (size_t)cB * cG * cN * cDG;
  _Float16* vTb   = wsb + off; off += (size_t)cB * cF * cN;
  _Float16* sbuf  = wsb + off; off += (size_t)cB * cG * cN * cN;
  _Float16* pbias = wsb + off; off += (size_t)cB * cG * cN * cN;
  (void)ws_size; (void)n_in; (void)in_sizes; (void)out_size;

  // ---- 1. precision conversion ----
  auto cvt = [&](const float* src, _Float16* dst, int n) {
    cvt_f32_to_f16<<<(n + 255) / 256, 256, 0, stream>>>(src, dst, n);
  };
  cvt(roi, Xh,  cBN * cF);
  cvt(Wq,  Wqh, cD * cF);
  cvt(Wk,  Wkh, cD * cF);
  cvt(Wv,  Wvh, cD * cF);
  cvt(Wo,  Woh, cG * cDG * cF);

  // ---- 2. position bias: stream pe once, all 16 groups per pass ----
  pbias_kernel<<<cB * cN, cN, 0, stream>>>(pe, Wp, bp, pbias);

  // ---- 3. QKV projections (WMMA) ----
  const int qkv_tiles  = (cBN / 16) * (cD / 16);  // 16384 tiles
  const int qkv_blocks = qkv_tiles / 4;           // 4 waves/block
  const float inv_sqrt_dg = 0.125f;               // 1/sqrt(64)
  qkv_gemm<<<qkv_blocks, 128, 0, stream>>>(Xh, Wqh, bq, qbuf, 0, inv_sqrt_dg);
  qkv_gemm<<<qkv_blocks, 128, 0, stream>>>(Xh, Wkh, bk, kbuf, 1, 1.0f);
  qkv_gemm<<<qkv_blocks, 128, 0, stream>>>(Xh, Wvh, bv, vTb,  2, 1.0f);

  // ---- 4. scores + bias + softmax (WMMA qk) ----
  scores_kernel<<<cB * cG * (cN / 16), 32, 0, stream>>>(qbuf, kbuf, pbias, sbuf);

  // ---- 5. S@V then grouped output projection (WMMA x2, fused) ----
  out_kernel<<<cB * cG * (cN / 16), 128, 0, stream>>>(sbuf, vTb, Woh, bo, out);
}
